// EquivairantMultiheadAttention_4063039062481
// MI455X (gfx1250) — matrix operand order repaired and checked by reading, not compiled, not run on hardware
//
#include <hip/hip_runtime.h>
#include <hip/hip_bf16.h>

// ---------------- problem constants ----------------
#define Bb   2
#define Nn   2048
#define BN   (Bb*Nn)        // 4096 rows
#define Mm   64
#define Hh   8
#define Dd   32
#define Cc   256
#define POSd 6

typedef __attribute__((ext_vector_type(16))) _Float16 f16x16;
typedef __attribute__((ext_vector_type(8)))  _Float16 h8;
typedef __attribute__((ext_vector_type(4)))  _Float16 h4;
typedef __attribute__((ext_vector_type(8)))  float    f32x8;

// ---------------- helpers ----------------
__device__ __forceinline__ float wred_sum(float x) {
#pragma unroll
  for (int off = 16; off; off >>= 1) x += __shfl_xor(x, off, 32);
  return x;
}
__device__ __forceinline__ float wred_max(float x) {
#pragma unroll
  for (int off = 16; off; off >>= 1) x = fmaxf(x, __shfl_xor(x, off, 32));
  return x;
}
__device__ __forceinline__ f16x16 cat8(h8 a, h8 b) {
  f16x16 r;
#pragma unroll
  for (int i = 0; i < 8; ++i) { r[i] = a[i]; r[i + 8] = b[i]; }
  return r;
}

// ---------------- f32 -> f16 cast (once per buffer; GEMM loop stays conversion-free) ----
__global__ void __launch_bounds__(256)
cast_f16_kernel(const float* __restrict__ src, _Float16* __restrict__ dst, int n) {
  const int i4 = (blockIdx.x * 256 + threadIdx.x) * 4;
  if (i4 < n) {
    float4 x = *(const float4*)(src + i4);
    h4 y;
    y[0] = (_Float16)x.x; y[1] = (_Float16)x.y;
    y[2] = (_Float16)x.z; y[3] = (_Float16)x.w;
    *(h4*)(dst + i4) = y;
  }
}

// ---------------- WMMA GEMM: C(4096x256) = A16(4096x256) * W16(256x256)^T + bias ----------
// Per-wave 16x16 output tile; block = 8 waves; wave w covers n-tiles {2w, 2w+1},
// reusing its A fragment. K swept in 8 steps of 32 (fully unrolled -> 16 v_wmma/wave).
// Operands pre-cast to f16: inner loop is 6x b128 loads + 2x v_wmma per step.
__global__ void __launch_bounds__(256)
gemm256_wmma_kernel(const _Float16* __restrict__ A, const _Float16* __restrict__ W,
                    const float* __restrict__ bias, float* __restrict__ C) {
  const int Kk = Cc, Ncol = Cc;
  const int mtile = blockIdx.x;            // 16-row tile of A/C
  const int w     = threadIdx.x >> 5;      // wave id 0..7
  const int lane  = threadIdx.x & 31;
  const int grp   = lane >> 4;             // 0/1 (lane group)
  const int l16   = lane & 15;

  const _Float16* arow  = A + ((size_t)mtile * 16 + l16) * Kk;
  const int nt0 = 2 * w, nt1 = 2 * w + 1;
  const _Float16* wrow0 = W + ((size_t)nt0 * 16 + l16) * Kk;  // B[k][n] = W[n][k]
  const _Float16* wrow1 = W + ((size_t)nt1 * 16 + l16) * Kk;

  f32x8 acc0 = {}; f32x8 acc1 = {};
#pragma unroll
  for (int k0 = 0; k0 < Kk; k0 += 32) {
    // A 16x32 frag: lane row = l16, k = k0 + 16*grp + i (16 contiguous halfs)
    f16x16 af = cat8(*(const h8*)(arow + k0 + 16 * grp),
                     *(const h8*)(arow + k0 + 16 * grp + 8));
    // B 32x16 frag: lane col = l16, k = k0 + 16*grp + i (16 contiguous halfs)
    f16x16 bf0 = cat8(*(const h8*)(wrow0 + k0 + 16 * grp),
                      *(const h8*)(wrow0 + k0 + 16 * grp + 8));
    f16x16 bf1 = cat8(*(const h8*)(wrow1 + k0 + 16 * grp),
                      *(const h8*)(wrow1 + k0 + 16 * grp + 8));
    acc0 = __builtin_amdgcn_wmma_f32_16x16x32_f16(false, af, false, bf0,
                                                  (short)0, acc0, false, false);
    acc1 = __builtin_amdgcn_wmma_f32_16x16x32_f16(false, af, false, bf1,
                                                  (short)0, acc1, false, false);
  }
  // C/D layout: VGPR r -> local row r + 8*grp, col = l16 within n-tile
#pragma unroll
  for (int r = 0; r < 8; ++r) {
    const int row = mtile * 16 + 8 * grp + r;
    const int c0  = nt0 * 16 + l16;
    const int c1  = nt1 * 16 + l16;
    C[(size_t)row * Ncol + c0] = acc0[r] + bias[c0];
    C[(size_t)row * Ncol + c1] = acc1[r] + bias[c1];
  }
}

// ---------------- fused neighborhood attention ----------------
// 1 block per node (b,n); wave w owns head h=w. Uses the t2 = pg.S + c rewrite
// (eliminates the (B,N,M,256) 'e' tensor). K/Vpre gathers are L2-resident (4 MB tables).
// Emits context directly as f16 for the output-projection GEMM.
__global__ void __launch_bounds__(256)
attn_kernel(const float* __restrict__ pg,          // (B,N,N,6)
            const int* __restrict__ nidx,          // (B,N,64)
            const unsigned char* __restrict__ mask,// (B,N) bool
            const float* __restrict__ Q,           // (B,N,256)
            const float* __restrict__ Kt,          // (B,N,256)
            const float* __restrict__ Vp,          // (B,N,256)
            const float* __restrict__ Wl,          // (256,6)
            const float* __restrict__ bl,          // (256)
            const float* __restrict__ uu,          // (8,32) flat
            const float* __restrict__ vv,          // (8,32) flat
            _Float16* __restrict__ ctx16)          // (B,N,256) f16
{
  __shared__ int   sidx[Mm];
  __shared__ float smask[Mm];
  __shared__ float sQ[Cc];
  __shared__ float sattn[Hh * Mm];

  const int bn = blockIdx.x;
  const int b  = bn / Nn;
  const int n  = bn % Nn;
  const int t  = threadIdx.x;

  if (t < Mm) {
    const int id = nidx[(size_t)bn * Mm + t];
    sidx[t]  = id;
    smask[t] = mask[(size_t)b * Nn + id] ? 1.0f : 0.0f;
  }
  sQ[t] = Q[(size_t)bn * Cc + t];
  __syncthreads();

  const int h    = t >> 5;          // head == wave
  const int lane = t & 31;
  const int cb   = h * Dd;          // channel base of this head
  const int c    = cb + lane;       // this lane's channel

  // ---- per-node head precompute: S[6], c_h, t3 (wave butterfly reductions) ----
  const float qvd = sQ[c] + vv[c];
  float S[POSd];
#pragma unroll
  for (int j = 0; j < POSd; ++j) S[j] = wred_sum(Wl[c * POSd + j] * qvd);
  const float ch = wred_sum(bl[c] * qvd);
  const float t3 = wred_sum(Kt[(size_t)bn * Cc + c] * uu[c]);

  // ---- scores: 2 neighbors per lane ----
  const float rscale = 0.17677669529663687f;  // 1/sqrt(32)
  float sc[2];
#pragma unroll
  for (int r = 0; r < 2; ++r) {
    const int m  = lane + 32 * r;
    const int id = sidx[m];
    const float* krow = Kt + ((size_t)b * Nn + id) * Cc + cb;   // 128B, L2-resident
    __builtin_prefetch(Vp + ((size_t)b * Nn + id) * Cc + cb, 0, 1);
    float dot = 0.f;
#pragma unroll
    for (int d = 0; d < Dd; d += 4) {
      float4 kv = *(const float4*)(krow + d);
      dot += kv.x * sQ[cb + d]     + kv.y * sQ[cb + d + 1]
           + kv.z * sQ[cb + d + 2] + kv.w * sQ[cb + d + 3];
    }
    const float* pgrow = pg + (((size_t)b * Nn + n) * Nn + id) * POSd;
    float t2 = ch;
#pragma unroll
    for (int j = 0; j < POSd; ++j) t2 += pgrow[j] * S[j];
    const float s = (dot + t2 + t3) * rscale;
    sc[r] = (smask[m] != 0.0f) ? s : -1e38f;
  }

  // ---- softmax over 64 neighbors (2 values/lane) ----
  const float mx = wred_max(fmaxf(sc[0], sc[1]));
  const float e0 = __expf(sc[0] - mx);
  const float e1 = __expf(sc[1] - mx);
  const float inv = 1.0f / wred_sum(e0 + e1);
  sattn[h * Mm + lane]      = e0 * inv;   // same-wave LDS: DS ops in-order
  sattn[h * Mm + lane + 32] = e1 * inv;

  // ---- aggregate values: lane = output channel e ----
  const float* attn = &sattn[h * Mm];
  float acc = 0.f;
#pragma unroll 8
  for (int m = 0; m < Mm; ++m) {
    const int id = sidx[m];
    acc = fmaf(attn[m], Vp[((size_t)b * Nn + id) * Cc + cb + lane], acc);
  }
  ctx16[(size_t)bn * Cc + cb + lane] = (_Float16)acc;
}

// ---------------- launcher ----------------
extern "C" void kernel_launch(void* const* d_in, const int* in_sizes, int n_in,
                              void* d_out, int out_size, void* d_ws, size_t ws_size,
                              hipStream_t stream) {
  const float*         pg   = (const float*)d_in[0];
  const float*         cf   = (const float*)d_in[1];
  const unsigned char* mask = (const unsigned char*)d_in[2];
  const int*           nidx = (const int*)d_in[3];
  const float* Wq  = (const float*)d_in[4];
  const float* bq  = (const float*)d_in[5];
  const float* Wk  = (const float*)d_in[6];
  const float* bk  = (const float*)d_in[7];
  const float* Wl  = (const float*)d_in[8];
  const float* bl  = (const float*)d_in[9];
  const float* uu  = (const float*)d_in[10];
  const float* vvp = (const float*)d_in[11];
  const float* Win = (const float*)d_in[12];
  const float* bin = (const float*)d_in[13];
  const float* Wout= (const float*)d_in[14];
  const float* bout= (const float*)d_in[15];
  (void)in_sizes; (void)n_in; (void)out_size; (void)ws_size;

  float* out = (float*)d_out;
  const size_t per = (size_t)BN * Cc;       // 1,048,576 elems
  const int    wsz = Cc * Cc;               // 65,536 weight elems

  float*    Qb     = (float*)d_ws;
  float*    Kb     = Qb + per;
  float*    Vb     = Kb + per;
  _Float16* cf16   = (_Float16*)(Vb + per);
  _Float16* ctx16  = cf16 + per;
  _Float16* Wq16   = ctx16 + per;
  _Float16* Wk16   = Wq16 + wsz;
  _Float16* Win16  = Wk16 + wsz;
  _Float16* Wout16 = Win16 + wsz;

  const dim3 blk(256);
  // one-time f32 -> f16 casts (activations shared by 3 GEMMs, weights by all blocks)
  cast_f16_kernel<<<dim3((int)per / 1024), blk, 0, stream>>>(cf,   cf16,  (int)per);
  cast_f16_kernel<<<dim3(wsz / 1024),      blk, 0, stream>>>(Wq,   Wq16,  wsz);
  cast_f16_kernel<<<dim3(wsz / 1024),      blk, 0, stream>>>(Wk,   Wk16,  wsz);
  cast_f16_kernel<<<dim3(wsz / 1024),      blk, 0, stream>>>(Win,  Win16, wsz);
  cast_f16_kernel<<<dim3(wsz / 1024),      blk, 0, stream>>>(Wout, Wout16, wsz);

  const dim3 ggrid(BN / 16);                // 256 blocks of 16 rows
  gemm256_wmma_kernel<<<ggrid, blk, 0, stream>>>(cf16, Wq16, bq, Qb);
  gemm256_wmma_kernel<<<ggrid, blk, 0, stream>>>(cf16, Wk16, bk, Kb);
  gemm256_wmma_kernel<<<ggrid, blk, 0, stream>>>(cf16, Win16, bin, Vb);
  attn_kernel<<<dim3(BN), blk, 0, stream>>>(pg, nidx, mask, Qb, Kb, Vb,
                                            Wl, bl, uu, vvp, ctx16);
  gemm256_wmma_kernel<<<ggrid, blk, 0, stream>>>(ctx16, Wout16, bout, out);
}
